// self_attn_enc_40200893890895
// MI455X (gfx1250) — matrix operand order repaired and checked by reading, not compiled, not run on hardware
//
#include <hip/hip_runtime.h>

typedef _Float16 half_t;
typedef __attribute__((ext_vector_type(16))) _Float16 v16h;
typedef __attribute__((ext_vector_type(8)))  _Float16 v8h;
typedef __attribute__((ext_vector_type(8)))  float    v8f;

#define N_NODES 8192
#define D 128
#define D_PREV 122

// ---- CDNA5 WMMA fragment loaders (layouts per CDNA5 ISA §7.12.2) ----

// 16-bit A matrix 16x32: lane L holds M=L%16; K = h + (h>=8?8:0) + (L>=16?8:0).
__device__ __forceinline__ v16h load_frag_a(const half_t* base, int ld, int lane) {
  const int row = lane & 15, lg = (lane >> 4) & 1;
  const half_t* p = base + row * ld + lg * 8;
  v8h lo = *(const v8h*)(p);
  v8h hi = *(const v8h*)(p + 16);
  return __builtin_shufflevector(lo, hi, 0,1,2,3,4,5,6,7,8,9,10,11,12,13,14,15);
}

// 16-bit B matrix 32x16: lane L holds N=L%16; K = h + (h>=8?8:0) + (L>=16?8:0).
__device__ __forceinline__ v16h load_frag_b(const half_t* base, int ld, int lane) {
  const int col = lane & 15, lg = (lane >> 4) & 1;
  const half_t* p = base + col * ld + lg * 8;
  v8h lo = *(const v8h*)(p);
  v8h hi = *(const v8h*)(p + 16);
  return __builtin_shufflevector(lo, hi, 0,1,2,3,4,5,6,7,8,9,10,11,12,13,14,15);
}

__device__ __forceinline__ v8f wmma_f16(v16h a, v16h b, v8f c) {
  return __builtin_amdgcn_wmma_f32_16x16x32_f16(false, a, false, b, (short)0, c,
                                                false, false);
}

#define WAIT_ASYNC() asm volatile("s_wait_asynccnt 0x0" ::: "memory")
#define WAIT_DS()    asm volatile("s_wait_dscnt 0x0" ::: "memory")

// Per-lane 16B async copy global->LDS (gfx1250 GLOBAL_LOAD_ASYNC_TO_LDS_B128,
// ASYNCcnt-tracked; LDS address is the wave-relative low-32 of the generic ptr).
__device__ __forceinline__ void async_cp16(uint32_t lds, uint64_t gaddr) {
  asm volatile("global_load_async_to_lds_b128 %0, %1, off"
               :: "v"(lds), "v"(gaddr) : "memory");
}

// ---- kernel 1: f32 -> f16 weight conversion (+ Wv split into Wv_q / Wv_k) ----
__global__ void prep_weights(const float* __restrict__ Wq, const float* __restrict__ Wk,
                             const float* __restrict__ Wv,
                             half_t* __restrict__ whq, half_t* __restrict__ whk,
                             half_t* __restrict__ whvq, half_t* __restrict__ whvk) {
  int i = blockIdx.x * blockDim.x + threadIdx.x;
  if (i < D * D) {
    whq[i] = (half_t)Wq[i];
    whk[i] = (half_t)Wk[i];
    int o = i >> 7, d = i & 127;
    whvq[i] = (half_t)Wv[o * 2 * D + d];
    whvk[i] = (half_t)Wv[o * 2 * D + D + d];
  }
}

// ---- kernel 2: x = concat(nodes,centre,prev); Q,K projections; vq^T; K@Wvk^T+bv ----
__global__ __launch_bounds__(128) void stage1(
    const float* __restrict__ nodes, const float* __restrict__ centre,
    const float* __restrict__ prev,
    const float* __restrict__ bq, const float* __restrict__ bk,
    const float* __restrict__ bv,
    const half_t* __restrict__ whq, const half_t* __restrict__ whk,
    const half_t* __restrict__ whvq, const half_t* __restrict__ whvk,
    float* __restrict__ keysf, half_t* __restrict__ keysh,
    half_t* __restrict__ queriesh, half_t* __restrict__ vqT,
    float* __restrict__ kvb) {
  __shared__ __align__(16) half_t xb[4][16][136];
  __shared__ __align__(16) half_t qb[4][16][136];
  const int wave = threadIdx.x >> 5;
  const int lane = threadIdx.x & 31;
  const int row = lane & 15, lg = (lane >> 4) & 1;
  const int i0 = (blockIdx.x * 4 + wave) * 16;

  for (int idx = lane; idx < 16 * D; idx += 32) {
    int r = idx >> 7, c = idx & 127;
    float v;
    if (c < 3)      v = nodes[(i0 + r) * 3 + c];
    else if (c < 6) v = centre[c - 3];
    else            v = prev[(i0 + r) * D_PREV + (c - 6)];
    xb[wave][r][c] = (half_t)v;
  }
  __syncthreads();

  v16h xa[4];
  for (int dk = 0; dk < 4; ++dk)
    xa[dk] = load_frag_a(&xb[wave][0][dk * 32], 136, lane);

  // queries = x @ Wq^T + bq
  for (int c = 0; c < 8; ++c) {
    v8f acc = {};
    for (int dk = 0; dk < 4; ++dk) {
      v16h b = load_frag_b(whq + (c * 16) * D + dk * 32, D, lane);
      acc = wmma_f16(xa[dk], b, acc);
    }
    float bias = bq[c * 16 + row];
    for (int r = 0; r < 8; ++r) {
      float q = acc[r] + bias;
      queriesh[(i0 + r + 8 * lg) * D + c * 16 + row] = (half_t)q;
      qb[wave][r + 8 * lg][c * 16 + row] = (half_t)q;
    }
  }
  __syncthreads();

  // vq = queries @ Wvq^T, stored transposed (vqT[o][n])
  v16h qa[4];
  for (int dk = 0; dk < 4; ++dk)
    qa[dk] = load_frag_a(&qb[wave][0][dk * 32], 136, lane);
  for (int c = 0; c < 8; ++c) {
    v8f acc = {};
    for (int dk = 0; dk < 4; ++dk) {
      v16h b = load_frag_b(whvq + (c * 16) * D + dk * 32, D, lane);
      acc = wmma_f16(qa[dk], b, acc);
    }
    for (int r = 0; r < 8; ++r)
      vqT[(size_t)(c * 16 + row) * N_NODES + i0 + r + 8 * lg] = (half_t)acc[r];
  }

  // keys = x @ Wk^T + bk
  for (int c = 0; c < 8; ++c) {
    v8f acc = {};
    for (int dk = 0; dk < 4; ++dk) {
      v16h b = load_frag_b(whk + (c * 16) * D + dk * 32, D, lane);
      acc = wmma_f16(xa[dk], b, acc);
    }
    float bias = bk[c * 16 + row];
    for (int r = 0; r < 8; ++r) {
      float k = acc[r] + bias;
      int gi = (i0 + r + 8 * lg) * D + c * 16 + row;
      keysf[gi] = k;
      keysh[gi] = (half_t)k;
      qb[wave][r + 8 * lg][c * 16 + row] = (half_t)k;
    }
  }
  __syncthreads();

  // kvb = keys @ Wvk^T + bv
  v16h kf[4];
  for (int dk = 0; dk < 4; ++dk)
    kf[dk] = load_frag_a(&qb[wave][0][dk * 32], 136, lane);
  for (int c = 0; c < 8; ++c) {
    v8f acc = {};
    for (int dk = 0; dk < 4; ++dk) {
      v16h b = load_frag_b(whvk + (c * 16) * D + dk * 32, D, lane);
      acc = wmma_f16(kf[dk], b, acc);
    }
    float bias = bv[c * 16 + row];
    for (int r = 0; r < 8; ++r)
      kvb[(i0 + r + 8 * lg) * D + c * 16 + row] = acc[r] + bias;
  }
}

// ---- kernel 3: flash attention with async-LDS double-buffered operand staging ----
// Block = 128 threads = 4 waves x 16 rows. Per 32-col chunk the block stages
// 16 KB: queriesh rows j0..j0+31 (8 KB, contiguous) + vqT[:, j0..j0+31] (8 KB).
__global__ __launch_bounds__(128) void attn(
    const half_t* __restrict__ keysh, const half_t* __restrict__ queriesh,
    const half_t* __restrict__ vqT, const float* __restrict__ keysf,
    const float* __restrict__ kvb, float* __restrict__ out) {
  __shared__ __align__(16) half_t stage[2][8192];  // [buf][32x128 q | 128x32 v]
  __shared__ __align__(16) half_t pb[4][16][40];   // per-wave P transpose staging
  const int tid = threadIdx.x;
  const int wave = tid >> 5;
  const int lane = tid & 31;
  const int row = lane & 15, lg = (lane >> 4) & 1;
  const int i0 = (blockIdx.x * 4 + wave) * 16;

  const uint32_t lds_stage0 = (uint32_t)(uintptr_t)(&stage[0][0]);
  const char* qbase0 = (const char*)queriesh;
  const char* vbase0 = (const char*)vqT;

  v16h ka[4];
  for (int dk = 0; dk < 4; ++dk)
    ka[dk] = load_frag_a(keysh + i0 * D + dk * 32, D, lane);

  float m[8], lp[8];
  v8f o[8];
  for (int r = 0; r < 8; ++r) { m[r] = -__builtin_inff(); lp[r] = 0.f; }
  for (int c = 0; c < 8; ++c) { v8f z = {}; o[c] = z; }

  // issue prefetch of chunk 0 into buffer 0 (8 x 16B per thread)
  {
    const char* qb_ = qbase0;
    #pragma unroll
    for (int it = 0; it < 8; ++it) {
      int off = it * 2048 + tid * 16;
      uint64_t g;
      if (it < 4) g = (uint64_t)(uintptr_t)(qb_ + off);
      else {
        int rel = off - 8192;
        g = (uint64_t)(uintptr_t)(vbase0 + (size_t)(rel >> 6) * (N_NODES * 2) + (rel & 63));
      }
      async_cp16(lds_stage0 + (uint32_t)off, g);
    }
  }

  const int NK = N_NODES / 32;
  for (int k = 0; k < NK; ++k) {
    const int b = k & 1;
    WAIT_DS();
    WAIT_ASYNC();
    __syncthreads();  // buffer b ready for all; buffer b^1 free for refill

    if (k + 1 < NK) {  // prefetch next chunk into the other buffer
      const int j1 = (k + 1) * 32;
      const uint32_t ldsb = lds_stage0 + (uint32_t)((b ^ 1) * 16384);
      const char* qb_ = qbase0 + (size_t)j1 * (D * 2);
      const char* vb_ = vbase0 + (size_t)j1 * 2;
      #pragma unroll
      for (int it = 0; it < 8; ++it) {
        int off = it * 2048 + tid * 16;
        uint64_t g;
        if (it < 4) g = (uint64_t)(uintptr_t)(qb_ + off);
        else {
          int rel = off - 8192;
          g = (uint64_t)(uintptr_t)(vb_ + (size_t)(rel >> 6) * (N_NODES * 2) + (rel & 63));
        }
        async_cp16(ldsb + (uint32_t)off, g);
      }
    }

    const half_t* qs = &stage[b][0];     // [32][128]
    const half_t* vs = &stage[b][4096];  // [128][32]

    // S tiles: (lane,r) = S[i0 + r + 8lg][j0(+16) + row]
    v8f s0 = {}, s1 = {};
    for (int dk = 0; dk < 4; ++dk) {
      v16h b0 = load_frag_b(qs + dk * 32, D, lane);
      v16h b1 = load_frag_b(qs + 16 * D + dk * 32, D, lane);
      s0 = wmma_f16(ka[dk], b0, s0);
      s1 = wmma_f16(ka[dk], b1, s1);
    }

    // online softmax: row max via 16-lane xor reduction; lazy rescale
    float sc[8];
    int newmax = 0;
    for (int r = 0; r < 8; ++r) {
      float mc = fmaxf(s0[r], s1[r]);
      for (int off = 8; off >= 1; off >>= 1)
        mc = fmaxf(mc, __shfl_xor(mc, off, 32));
      float mn = fmaxf(m[r], mc);
      sc[r] = __expf(m[r] - mn);
      newmax |= (mn != m[r]);
      m[r] = mn;
    }
    if (__any(newmax)) {
      for (int r = 0; r < 8; ++r) {
        lp[r] *= sc[r];
        for (int c = 0; c < 8; ++c) o[c][r] *= sc[r];
      }
    }
    for (int r = 0; r < 8; ++r) {
      float p0 = __expf(s0[r] - m[r]);
      float p1 = __expf(s1[r] - m[r]);
      lp[r] += p0 + p1;  // per-lane partial row sum (reduced once at the end)
      pb[wave][r + 8 * lg][row]      = (half_t)p0;
      pb[wave][r + 8 * lg][16 + row] = (half_t)p1;
    }

    WAIT_DS();  // per-wave P LDS round trip (DS is in-order per wave)
    v16h pa = load_frag_a(&pb[wave][0][0], 40, lane);
    for (int c = 0; c < 8; ++c) {
      v16h bf = load_frag_b(vs + (c * 16) * 32, 32, lane);
      o[c] = wmma_f16(pa, bf, o[c]);
    }
  }

  // finalize: reduce per-lane partial sums across the 16-lane row group
  float l[8];
  for (int r = 0; r < 8; ++r) {
    float lv = lp[r];
    for (int off = 8; off >= 1; off >>= 1)
      lv += __shfl_xor(lv, off, 32);
    l[r] = lv;
  }
  for (int c = 0; c < 8; ++c)
    for (int r = 0; r < 8; ++r) {
      int gi = (i0 + r + 8 * lg) * D + c * 16 + row;
      out[gi] = keysf[gi] + kvb[gi] + o[c][r] / l[r];
    }
}

extern "C" void kernel_launch(void* const* d_in, const int* in_sizes, int n_in,
                              void* d_out, int out_size, void* d_ws, size_t ws_size,
                              hipStream_t stream) {
  const float* nodes  = (const float*)d_in[0];
  const float* centre = (const float*)d_in[1];
  const float* prev   = (const float*)d_in[2];
  const float* Wq     = (const float*)d_in[3];
  const float* bq     = (const float*)d_in[4];
  const float* Wk     = (const float*)d_in[5];
  const float* bk     = (const float*)d_in[6];
  const float* Wv     = (const float*)d_in[7];
  const float* bv     = (const float*)d_in[8];
  float* out = (float*)d_out;

  char* ws = (char*)d_ws;
  float*  keysf    = (float*)ws;  ws += (size_t)N_NODES * D * sizeof(float);
  float*  kvb      = (float*)ws;  ws += (size_t)N_NODES * D * sizeof(float);
  half_t* keysh    = (half_t*)ws; ws += (size_t)N_NODES * D * sizeof(half_t);
  half_t* queriesh = (half_t*)ws; ws += (size_t)N_NODES * D * sizeof(half_t);
  half_t* vqT      = (half_t*)ws; ws += (size_t)N_NODES * D * sizeof(half_t);
  half_t* whq      = (half_t*)ws; ws += (size_t)D * D * sizeof(half_t);
  half_t* whk      = (half_t*)ws; ws += (size_t)D * D * sizeof(half_t);
  half_t* whvq     = (half_t*)ws; ws += (size_t)D * D * sizeof(half_t);
  half_t* whvk     = (half_t*)ws; ws += (size_t)D * D * sizeof(half_t);

  prep_weights<<<(D * D + 255) / 256, 256, 0, stream>>>(Wq, Wk, Wv,
                                                        whq, whk, whvq, whvk);
  stage1<<<N_NODES / 64, 128, 0, stream>>>(nodes, centre, prev, bq, bk, bv,
                                           whq, whk, whvq, whvk,
                                           keysf, keysh, queriesh, vqT, kvb);
  attn<<<N_NODES / 64, 128, 0, stream>>>(keysh, queriesh, vqT, keysf, kvb, out);
}
